// GeometricIGBundleAdapter_63642825392413
// MI455X (gfx1250) — matrix operand order repaired and checked by reading, not compiled, not run on hardware
//
#include <hip/hip_runtime.h>
#include <hip/hip_bf16.h>
#include <math.h>

// ---------------------------------------------------------------------------
// GeometricIGBundleAdapter for MI455X (gfx1250, wave32, WMMA)
// Live dataflow (dead projections Wl/Wm/Ws/Wu eliminated):
//   h_bot = x@Wi+bi ; base = clip(h_bot@Wb+bb) ; fiber = h_bot@Wf+bf
//   wlog = h_bot@Ww+bw ; softmax ; transport 0.9*base[t]+0.1*base[t-1]
//   combined = [agg_coords | agg_sections] ; out = x + combined@Wo + bo
// GEMMs: v_wmma_f32_16x16x32_bf16, native v_cvt fp32->bf16, compile-time dims,
// double-buffered LDS (one barrier per 64-deep k-stage, 16 WMMAs per stage).
// ---------------------------------------------------------------------------

typedef __bf16 v16bf __attribute__((ext_vector_type(16)));
typedef __bf16 v8bf  __attribute__((ext_vector_type(8)));
typedef float  v8f   __attribute__((ext_vector_type(8)));

__device__ __forceinline__ __bf16 f2bf(float f) {
  return (__bf16)f;   // native RNE v_cvt on gfx1250
}

#define BM  128
#define BN  128
#define BKT 64    // staged k-depth (2 WMMA k-steps of 32)
#define LDSA 80   // LDS row stride in halves: 160B rows, 16B-aligned frags
#define LDSB 80

// EPI: 0 = +bias ; 1 = +bias then clip(-10,10) ; 2 = +bias + residual
template <int EPI, int K, int LDA, int LDB, int LDC>
__global__ __launch_bounds__(256)
void gemm_wmma_bf16(const float* __restrict__ A, const float* __restrict__ B,
                    const float* __restrict__ bias, const float* __restrict__ res,
                    float* __restrict__ C) {
  __shared__ __bf16 As[2][BM * LDSA];      // [m][k] row-major
  __shared__ __bf16 Bs[2][BN * LDSB];      // transposed: [n][k]

  const int tid   = threadIdx.x;
  const int lane  = tid & 31;
  const int wave  = tid >> 5;
  const int waveM = wave >> 1;             // 0..3 -> 32-row strip
  const int waveN = wave & 1;              // 0..1 -> 64-col strip
  const int lm    = lane & 15;
  const int lh    = lane >> 4;

  const int m0 = blockIdx.y * BM;
  const int n0 = blockIdx.x * BN;

  // ---- accumulators: 2 (M) x 4 (N) tiles of 16x16 ----
  v8f acc[2][4];
#pragma unroll
  for (int mi = 0; mi < 2; ++mi)
#pragma unroll
    for (int ni = 0; ni < 4; ++ni)
      acc[mi][ni] = (v8f){0.f, 0.f, 0.f, 0.f, 0.f, 0.f, 0.f, 0.f};

  // ---- loader index precompute (256 threads, 32 elems each per tile) ----
  const int am = tid >> 1;                 // A: row within tile (0..127)
  const int ak = (tid & 1) * 32;           // A: k-offset (0 or 32)
  const int bn = tid & 127;                // B: col within tile (0..127)
  const int bk = (tid >> 7) * 32;          // B: k-offset (0 or 32)

  const float* gA = A + (size_t)(m0 + am) * LDA + ak;
  const float* gB = B + (size_t)bk * LDB + n0 + bn;

  v8bf ra[4], rb[4];

  auto loadA = [&](int k0) {
    const float* gp = gA + k0;             // 16B aligned (LDA % 4 == 0)
#pragma unroll
    for (int q = 0; q < 4; ++q) {
      float4 u = ((const float4*)gp)[2 * q];
      float4 v = ((const float4*)gp)[2 * q + 1];
      ra[q] = (v8bf){f2bf(u.x), f2bf(u.y), f2bf(u.z), f2bf(u.w),
                     f2bf(v.x), f2bf(v.y), f2bf(v.z), f2bf(v.w)};
    }
  };
  auto loadB = [&](int k0) {
    const float* gp = gB + (size_t)k0 * LDB;
#pragma unroll
    for (int q = 0; q < 4; ++q)
#pragma unroll
      for (int j = 0; j < 8; ++j)
        rb[q][j] = f2bf(gp[(size_t)(q * 8 + j) * LDB]);   // const offsets
  };
  auto stage = [&](int buf) {
    __bf16* sa = &As[buf][am * LDSA + ak];
    __bf16* sb = &Bs[buf][bn * LDSB + bk];
#pragma unroll
    for (int q = 0; q < 4; ++q) {
      *(v8bf*)(sa + 8 * q) = ra[q];
      *(v8bf*)(sb + 8 * q) = rb[q];
    }
  };

  // ---- prologue: stage tile 0 ----
  loadA(0);
  loadB(0);
  stage(0);
  int cur = 0;

  for (int k0 = 0; k0 < K; k0 += BKT) {
    __syncthreads();
    const bool has_next = (k0 + BKT) < K;
    if (has_next) { loadA(k0 + BKT); loadB(k0 + BKT); }  // overlap with WMMAs

    const __bf16* Ab = As[cur];
    const __bf16* Bb = Bs[cur];
#pragma unroll
    for (int kk = 0; kk < BKT; kk += 32) {
      // A 16x32 frag: e<8 -> K=8*lh+e ; e>=8 -> K=16+8*lh+(e-8)
      v16bf af[2];
#pragma unroll
      for (int mi = 0; mi < 2; ++mi) {
        const __bf16* pa = Ab + (waveM * 32 + mi * 16 + lm) * LDSA + kk;
        v8bf lo = *(const v8bf*)(pa + 8 * lh);
        v8bf hi = *(const v8bf*)(pa + 16 + 8 * lh);
        af[mi] = __builtin_shufflevector(lo, hi, 0, 1, 2, 3, 4, 5, 6, 7,
                                         8, 9, 10, 11, 12, 13, 14, 15);
      }
      // B 32x16 frag: e<8 -> K=8*lh+e ; e>=8 -> K=16+8*lh+(e-8), N=lm  (Bs is [n][k])
      v16bf bfr[4];
#pragma unroll
      for (int ni = 0; ni < 4; ++ni) {
        const __bf16* pb = Bb + (waveN * 64 + ni * 16 + lm) * LDSB + kk;
        v8bf lo = *(const v8bf*)(pb + 8 * lh);
        v8bf hi = *(const v8bf*)(pb + 16 + 8 * lh);
        bfr[ni] = __builtin_shufflevector(lo, hi, 0, 1, 2, 3, 4, 5, 6, 7,
                                          8, 9, 10, 11, 12, 13, 14, 15);
      }
#pragma unroll
      for (int mi = 0; mi < 2; ++mi)
#pragma unroll
        for (int ni = 0; ni < 4; ++ni)
          acc[mi][ni] = __builtin_amdgcn_wmma_f32_16x16x32_bf16(
              false, af[mi], false, bfr[ni], (short)0, acc[mi][ni], false, false);
    }

    if (has_next) { stage(cur ^ 1); cur ^= 1; }
  }

  // ---- epilogue: C/D layout -> VGPR r holds M = r + 8*lh, N = lm ----
#pragma unroll
  for (int mi = 0; mi < 2; ++mi)
#pragma unroll
    for (int ni = 0; ni < 4; ++ni) {
      const int col = n0 + waveN * 64 + ni * 16 + lm;
      const float bv = bias[col];
#pragma unroll
      for (int r = 0; r < 8; ++r) {
        const int row = m0 + waveM * 32 + mi * 16 + lh * 8 + r;
        float v = acc[mi][ni][r] + bv;
        if (EPI == 1) v = fminf(fmaxf(v, -10.0f), 10.0f);
        const size_t idx = (size_t)row * LDC + col;
        if (EPI == 2) v += res[idx];
        C[idx] = v;
      }
    }
}

// ---- w_logits: one wave per token; 1024-dot x 8 outputs, shfl reduction ----
__global__ __launch_bounds__(256)
void wlogits_kernel(const float* __restrict__ h, const float* __restrict__ Ww,
                    const float* __restrict__ bw, float* __restrict__ wlog,
                    int ntok) {
  const int wid  = (int)((blockIdx.x * blockDim.x + threadIdx.x) >> 5);
  const int lane = threadIdx.x & 31;
  if (wid >= ntok) return;
  const float* hp = h + (size_t)wid * 1024;
  float acc[8] = {0.f, 0.f, 0.f, 0.f, 0.f, 0.f, 0.f, 0.f};
  for (int j = lane; j < 1024; j += 32) {
    const float hv = hp[j];
    const float4* wr4 = (const float4*)(Ww + (size_t)j * 8);  // 32B aligned
    const float4 w0 = wr4[0], w1 = wr4[1];
    acc[0] += hv * w0.x; acc[1] += hv * w0.y;
    acc[2] += hv * w0.z; acc[3] += hv * w0.w;
    acc[4] += hv * w1.x; acc[5] += hv * w1.y;
    acc[6] += hv * w1.z; acc[7] += hv * w1.w;
  }
#pragma unroll
  for (int p = 0; p < 8; ++p)
#pragma unroll
    for (int off = 16; off > 0; off >>= 1)
      acc[p] += __shfl_xor(acc[p], off, 32);
  if (lane < 8) wlog[(size_t)wid * 8 + lane] = acc[lane] + bw[lane];
}

// ---- softmax + transport + aggregate -> combined[ntok][128] ----
__global__ __launch_bounds__(256)
void fuse_kernel(const float* __restrict__ base, const float* __restrict__ fiber,
                 const float* __restrict__ wlog, float* __restrict__ combined,
                 int ntok, int T) {
  const int wid  = (int)((blockIdx.x * blockDim.x + threadIdx.x) >> 5);
  const int lane = threadIdx.x & 31;
  if (wid >= ntok) return;
  const int t = wid % T;

  // softmax over P=8 (replicated per lane; trivial)
  float wl[8];
  float mx = -1e30f;
#pragma unroll
  for (int p = 0; p < 8; ++p) { wl[p] = wlog[(size_t)wid * 8 + p]; mx = fmaxf(mx, wl[p]); }
  float s = 0.f;
#pragma unroll
  for (int p = 0; p < 8; ++p) { wl[p] = __expf(wl[p] - mx); s += wl[p]; }
  const float inv = 1.0f / s;

  const float* bcur  = base + (size_t)wid * 512;
  const float* bprev = (t == 0) ? bcur : (bcur - 512);   // t==0: 0.9c+0.1c == c
  const float* fc    = fiber + (size_t)wid * 512;

  const int d0 = 2 * lane;                                // lane handles d0, d0+1
  float ac0 = 0.f, ac1 = 0.f, as0 = 0.f, as1 = 0.f;
#pragma unroll
  for (int p = 0; p < 8; ++p) {
    const float w = wl[p] * inv;
    const float2 c = *(const float2*)(bcur + p * 64 + d0);
    const float2 q = *(const float2*)(bprev + p * 64 + d0);
    const float2 f = *(const float2*)(fc + p * 64 + d0);
    ac0 += w * (0.9f * c.x + 0.1f * q.x);
    ac1 += w * (0.9f * c.y + 0.1f * q.y);
    as0 += w * f.x;
    as1 += w * f.y;
  }
  float* out = combined + (size_t)wid * 128;
  *(float2*)(out + d0)      = make_float2(ac0, ac1);
  *(float2*)(out + 64 + d0) = make_float2(as0, as1);
}

// ---------------------------------------------------------------------------
extern "C" void kernel_launch(void* const* d_in, const int* in_sizes, int n_in,
                              void* d_out, int out_size, void* d_ws, size_t ws_size,
                              hipStream_t stream) {
  (void)in_sizes; (void)n_in; (void)out_size; (void)ws_size;
  // setup_inputs order: x,Wi,bi,Wb,bb,Wf,bf,Wl,bl,Ww,bw,Wm,bm,Ws,bs,Wu,bu,Wo,bo
  const float* x  = (const float*)d_in[0];
  const float* Wi = (const float*)d_in[1];
  const float* bi = (const float*)d_in[2];
  const float* Wb = (const float*)d_in[3];
  const float* bb = (const float*)d_in[4];
  const float* Wf = (const float*)d_in[5];
  const float* bf = (const float*)d_in[6];
  const float* Ww = (const float*)d_in[9];
  const float* bw = (const float*)d_in[10];
  const float* Wo = (const float*)d_in[17];
  const float* bo = (const float*)d_in[18];
  float* out = (float*)d_out;

  const int NTOK = 4 * 2048;   // B*T = 8192
  const int Tlen = 2048;

  // workspace carve-out (~72 MB)
  char* ws = (char*)d_ws;
  size_t off = 0;
  float* h_bot = (float*)(ws + off); off += (size_t)NTOK * 1024 * sizeof(float);
  float* base  = (float*)(ws + off); off += (size_t)NTOK * 512 * sizeof(float);
  float* fiber = (float*)(ws + off); off += (size_t)NTOK * 512 * sizeof(float);
  float* wlog  = (float*)(ws + off); off += (size_t)NTOK * 8 * sizeof(float);
  float* comb  = (float*)(ws + off); off += (size_t)NTOK * 128 * sizeof(float);

  dim3 blk(256);
  const int gy = NTOK / BM;    // 64

  // 1) h_bot = x @ Wi + bi          [8192 x 1024, K=4096]
  gemm_wmma_bf16<0, 4096, 4096, 1024, 1024>
      <<<dim3(1024 / BN, gy), blk, 0, stream>>>(x, Wi, bi, nullptr, h_bot);

  // 2) base = clip(h_bot @ Wb + bb) [8192 x 512, K=1024]
  gemm_wmma_bf16<1, 1024, 1024, 512, 512>
      <<<dim3(512 / BN, gy), blk, 0, stream>>>(h_bot, Wb, bb, nullptr, base);

  // 3) fiber = h_bot @ Wf + bf      [8192 x 512, K=1024]
  gemm_wmma_bf16<0, 1024, 1024, 512, 512>
      <<<dim3(512 / BN, gy), blk, 0, stream>>>(h_bot, Wf, bf, nullptr, fiber);

  // 4) w_logits                     [8192 x 8]
  wlogits_kernel<<<dim3(NTOK / 8), blk, 0, stream>>>(h_bot, Ww, bw, wlog, NTOK);

  // 5) softmax + transport + aggregate -> combined [8192 x 128]
  fuse_kernel<<<dim3(NTOK / 8), blk, 0, stream>>>(base, fiber, wlog, comb, NTOK, Tlen);

  // 6) out = x + combined @ Wo + bo [8192 x 4096, K=128]
  gemm_wmma_bf16<2, 128, 128, 4096, 4096>
      <<<dim3(4096 / BN, gy), blk, 0, stream>>>(comb, Wo, bo, x, out);
}
